// ATTN_71305047048490
// MI455X (gfx1250) — matrix operand order repaired and checked by reading, not compiled, not run on hardware
//
#include <hip/hip_runtime.h>
#include <hip/hip_bf16.h>

// ---------------------------------------------------------------------------
// Temporal-graph attention, MI455X (gfx1250, wave32, WMMA + TDM).
//
// Algebraic folding (len_q == 1):
//   logits = C (W_k^T q)        -> kills the 134-GFLOP K projection
//   fc(out) = (attn C) (W_v fc) -> kills the 134-GFLOP V projection
// Remaining ~18 GFLOP; kernel is HBM-bound (~180 MB => ~8us @ 23.3 TB/s).
// Dense GEMMs: v_wmma_f32_16x16x32_bf16. Weights are pre-swizzled into
// fragment-ready layout so every B operand is two global_load_b128 per lane.
// h_dst tile is staged into LDS by the Tensor Data Mover (tensor_load_to_lds).
// ---------------------------------------------------------------------------

#define B_      8192
#define N_      20
#define E_      128
#define T_      64
#define F_      128
#define H_      4
#define QD_     192          // E + T
#define KD_     320          // E + T + F
#define HK_     1280         // H * KD
#define TB_     16           // batch rows per workgroup == one WMMA M tile
#define ROWS_   (TB_ * N_)   // 320 edge rows per workgroup
#define THREADS_ 256         // 8 waves (wave32)
#define LN_EPS_ 1e-5f

typedef __attribute__((ext_vector_type(16))) __bf16 v16bf;
typedef __attribute__((ext_vector_type(8)))  float  v8f;
typedef __attribute__((ext_vector_type(4)))  unsigned int v4u;
typedef __attribute__((ext_vector_type(8)))  int v8i;
typedef __attribute__((ext_vector_type(4)))  int v4i;

// ---- LDS layout (dynamic; 320KB/WGP on CDNA5) ------------------------------
#define OFF_C    0                      // C tile        320x320 bf16 = 204800
#define OFF_Q    204800                 // query tile    16x192 bf16  =   6144
#define OFF_B1   210944                 // q / ctx       16x1280 bf16 =  40960
#define OFF_B2   251904                 // wq_eff bf16, later x(f32)+HB+R
#define OFF_XB   OFF_B2                 // x residual    16x192 f32   =  12288
#define OFF_HB   (OFF_B2 + 12288)       // merge input   16x320 bf16  =  10240
#define OFF_R    (OFF_B2 + 22528)       // relu out      16x128 bf16  =   4096
#define OFF_S    292864                 // scores/attn   320x4 f32    =   5120
#define OFF_ST   297984                 // mu[16], rstd[16]           =    128
#define OFF_IDX  298112                 // nbr_idx tile  320 i32      =   1280
#define OFF_DT   299392                 // delta_t tile  320 f32      =   1280
#define OFF_HD   300672                 // h_dst tile (TDM) 16x128 f32 =  8192
#define SMEM_    308864

// ---- CDNA5 WMMA fragment helpers (ISA 7.12.2 layouts) ----------------------
// A 16x32 bf16 (MxK): lane&15 = M row; lane-halves split the K window.
static __device__ __forceinline__ v16bf load_A(const __bf16* base, int stride,
                                               int k0, int lane) {
  int m = lane & 15;
  int kh = (lane >> 4) * 16;
  v16bf a;
#pragma unroll
  for (int e = 0; e < 16; ++e) {
    int k = e + kh;
    a[e] = base[m * stride + k0 + k];
  }
  return a;
}
// B operand: fragment-ready weights -> one aligned 32B load per lane.
static __device__ __forceinline__ v16bf load_Bfrag(const __bf16* frag, int NT,
                                                   int kt, int nt, int lane) {
  return *(const v16bf*)(frag + ((((kt * NT) + nt) * 32) + lane) * 16);
}
// C/D f32 16x16: lane&15 = N; VGPR v: M = v + 8*(lane>>4)
static __device__ __forceinline__ void store_D_bf16(__bf16* dst, int stride,
                                                    int n0, v8f acc, int lane) {
  int n = n0 + (lane & 15);
  int mb = (lane >> 4) * 8;
#pragma unroll
  for (int v = 0; v < 8; ++v) dst[(mb + v) * stride + n] = (__bf16)acc[v];
}

static __device__ __forceinline__ v8f wmma_bf16(v16bf a, v16bf b, v8f c) {
  return __builtin_amdgcn_wmma_f32_16x16x32_bf16(false, a, false, b,
                                                 (short)0, c, false, false);
}

// ---------------------------------------------------------------------------
// Prep kernel 1: f32 -> bf16, swizzled into WMMA-B fragment layout:
//   frag[(kt*NT + nt)*32 + lane][16] ; lane&15 = n-in-tile,
//   lane>>4 selects K half; element e maps to K offset (lane>>4)*16 + e.
// ---------------------------------------------------------------------------
#define PW_T0 245760           // wq  : 6kt x 80nt x 512
#define PW_T1 655360           // wkT : 4 heads x (10kt x 20nt x 512)
#define PW_T2 696320           // mg1 : 10kt x 8nt x 512
#define PW_T3 712704           // mg2 : 4kt  x 8nt x 512
__global__ void prep_weights(const float* __restrict__ w_q,
                             const float* __restrict__ w_k,
                             const float* __restrict__ mg1_w,
                             const float* __restrict__ mg2_w,
                             __bf16* __restrict__ wq_f,
                             __bf16* __restrict__ wkT_f,
                             __bf16* __restrict__ mg1_f,
                             __bf16* __restrict__ mg2_f) {
  for (int i = blockIdx.x * blockDim.x + threadIdx.x; i < PW_T3;
       i += gridDim.x * blockDim.x) {
    if (i < PW_T0) {                     // wq [K=192][N=1280], NT=80
      int t = i >> 9, w = i & 511, lane = w >> 4, e = w & 15;
      int kt = t / 80, nt = t % 80;
      int n = nt * 16 + (lane & 15);
      int k = kt * 32 + ((lane >> 4) << 4) + e;
      wq_f[i] = (__bf16)w_q[k * HK_ + n];
    } else if (i < PW_T1) {              // wkT per head [K=d 320][N=c 320]
      int j = i - PW_T0;
      int h = j / 102400, jj = j % 102400;
      int t = jj >> 9, w = jj & 511, lane = w >> 4, e = w & 15;
      int kt = t / 20, nt = t % 20;
      int c = nt * 16 + (lane & 15);
      int d = kt * 32 + ((lane >> 4) << 4) + e;
      wkT_f[j] = (__bf16)w_k[c * HK_ + h * KD_ + d];
    } else if (i < PW_T2) {              // mg1 [K=320][N=128], NT=8
      int j = i - PW_T1;
      int t = j >> 9, w = j & 511, lane = w >> 4, e = w & 15;
      int kt = t / 8, nt = t % 8;
      int n = nt * 16 + (lane & 15);
      int k = kt * 32 + ((lane >> 4) << 4) + e;
      mg1_f[j] = (__bf16)mg1_w[k * E_ + n];
    } else {                             // mg2 [K=128][N=128], NT=8
      int j = i - PW_T2;
      int t = j >> 9, w = j & 511, lane = w >> 4, e = w & 15;
      int kt = t / 8, nt = t % 8;
      int n = nt * 16 + (lane & 15);
      int k = kt * 32 + ((lane >> 4) << 4) + e;
      mg2_f[j] = (__bf16)mg2_w[k * E_ + n];
    }
  }
}

// ---------------------------------------------------------------------------
// Prep kernel 2: Wvf[k=h*320+c][j] = sum_d w_v[c, h*320+d] * fc_w[h*320+d, j]
// written directly into fragment layout (K=1280 -> 40kt, N=192 -> 12nt).
// ---------------------------------------------------------------------------
__global__ void prep_wvf(const float* __restrict__ w_v,
                         const float* __restrict__ fc_w,
                         __bf16* __restrict__ wvf_f) {
  int i = blockIdx.x * blockDim.x + threadIdx.x;
  if (i >= HK_ * QD_) return;
  int k = i / QD_, j = i % QD_;
  int h = k / KD_, c = k % KD_;
  const float* wv = w_v + c * HK_ + h * KD_;
  const float* fw = fc_w + h * KD_ * QD_ + j;
  float acc = 0.f;
#pragma unroll 4
  for (int d = 0; d < KD_; ++d) acc += wv[d] * fw[d * QD_];
  // fragment position
  int kt = k >> 5, kk = k & 31;
  int lane = ((kk >> 4) << 4) | (j & 15);
  int e = kk & 15;
  int nt = j >> 4;
  wvf_f[((((kt * 12) + nt) * 32) + lane) * 16 + e] = (__bf16)acc;
}

// ---------------------------------------------------------------------------
// Main fused kernel: one workgroup == 16 batch rows (one WMMA M tile).
// ---------------------------------------------------------------------------
__global__ __launch_bounds__(THREADS_) void attn_fused(
    const float* __restrict__ h_dst, const float* __restrict__ src_h,
    const float* __restrict__ edge_feat, const float* __restrict__ delta_t,
    const float* __restrict__ time_w, const float* __restrict__ time_b,
    const float* __restrict__ fc_b, const float* __restrict__ ln_g,
    const float* __restrict__ ln_b, const float* __restrict__ mg1_b,
    const float* __restrict__ mg2_b, const int* __restrict__ nbr_idx,
    const __bf16* __restrict__ wq_f, const __bf16* __restrict__ wkT_f,
    const __bf16* __restrict__ wvf_f, const __bf16* __restrict__ mg1_f,
    const __bf16* __restrict__ mg2_f, float* __restrict__ out) {
  extern __shared__ char smem[];
  __bf16* Cl  = (__bf16*)(smem + OFF_C);
  __bf16* Ql  = (__bf16*)(smem + OFF_Q);
  __bf16* B1  = (__bf16*)(smem + OFF_B1);   // q bf16, later ctx bf16
  __bf16* WQE = (__bf16*)(smem + OFF_B2);
  float*  XB  = (float*)(smem + OFF_XB);    // aliases WQE (dead by then)
  __bf16* HB  = (__bf16*)(smem + OFF_HB);
  __bf16* Rl  = (__bf16*)(smem + OFF_R);
  float*  Sl  = (float*)(smem + OFF_S);
  float*  MU  = (float*)(smem + OFF_ST);
  float*  RS  = (float*)(smem + OFF_ST + 64);
  int*    IDX = (int*)(smem + OFF_IDX);
  float*  DT  = (float*)(smem + OFF_DT);
  float*  HD  = (float*)(smem + OFF_HD);    // h_dst tile, staged by TDM

  const int tid  = threadIdx.x;
  const int lane = tid & 31;
  const int wave = tid >> 5;
  const int b0   = blockIdx.x * TB_;

  // ---- Phase A0: stage per-tile gather metadata + h_dst tile --------------
  for (int i = tid; i < ROWS_; i += THREADS_) {
    IDX[i] = nbr_idx[b0 * N_ + i];
    DT[i]  = delta_t[b0 * N_ + i];
  }
#if defined(__has_builtin) && __has_builtin(__builtin_amdgcn_tensor_load_to_lds)
  if (wave == 0) {
    // Tensor DMA: 16x128 f32 h_dst tile == one contiguous 8KB block.
    // D# group0: count=1 | lds_addr | global_addr[56:0] | type=2.
    unsigned long long ga = (unsigned long long)(h_dst + (size_t)b0 * E_);
    v4u g0 = {1u, (unsigned)OFF_HD, (unsigned)(ga & 0xffffffffull),
              (unsigned)((ga >> 32) & 0x1ffffffull) | 0x80000000u};
    // D# group1: data_size=4B; tensor_dim0=2048, tile_dim0=2048 (1-D tile),
    // tensor_dim1=1, tile_dim1=0(unused), dim0_stride=2048.
    v8i g1 = {0x00020000, 0x08000000, 0x00010000, 0x08000000, 0, 2048, 0, 0};
    v4i z4 = {0, 0, 0, 0};
    v8i z8 = {0, 0, 0, 0, 0, 0, 0, 0};
    __builtin_amdgcn_tensor_load_to_lds(g0, g1, z4, z4, z8, 0);
    __builtin_amdgcn_s_wait_tensorcnt(0);
  }
#else
  for (int i = tid; i < TB_ * E_; i += THREADS_)
    HD[i] = h_dst[(size_t)b0 * E_ + i];
#endif
  __syncthreads();

  // ---- Phase A: build query tile and C tile (float4 gather + time enc) ----
  for (int i = tid; i < TB_ * QD_; i += THREADS_) {
    int m = i / QD_, j = i % QD_;
    float vq = (j < E_) ? HD[m * E_ + j] : cosf(time_b[j - E_]);
    Ql[i] = (__bf16)vq;
  }
  for (int i = tid; i < ROWS_ * (KD_ / 4); i += THREADS_) {
    int r = i / (KD_ / 4), c = (i % (KD_ / 4)) * 4;
    float4 f;
    if (c < E_) {
      f = *(const float4*)(src_h + (size_t)IDX[r] * E_ + c);
    } else if (c < E_ + F_) {
      f = *(const float4*)(edge_feat + ((size_t)b0 * N_ + r) * F_ + (c - E_));
    } else {
      int t = c - (E_ + F_);
      float dt = DT[r];
      f.x = cosf(dt * time_w[t + 0] + time_b[t + 0]);
      f.y = cosf(dt * time_w[t + 1] + time_b[t + 1]);
      f.z = cosf(dt * time_w[t + 2] + time_b[t + 2]);
      f.w = cosf(dt * time_w[t + 3] + time_b[t + 3]);
    }
    __bf16* dst = Cl + r * KD_ + c;
    dst[0] = (__bf16)f.x; dst[1] = (__bf16)f.y;
    dst[2] = (__bf16)f.z; dst[3] = (__bf16)f.w;
  }
  __syncthreads();

  // ---- Phase B: q = query @ w_q   [16x192]@[192x1280] -> B1 (bf16) --------
  for (int nt = wave; nt < 80; nt += 8) {
    __builtin_prefetch(wq_f + (size_t)nt * 512, 0, 1);  // global_prefetch_b8
    v8f acc = {};
#pragma unroll
    for (int kt = 0; kt < 6; ++kt) {
      v16bf a = load_A(Ql, QD_, kt * 32, lane);
      v16bf b = load_Bfrag(wq_f, 80, kt, nt, lane);
      acc = wmma_bf16(a, b, acc);
    }
    store_D_bf16(B1, HK_, nt * 16, acc, lane);
  }
  __syncthreads();

  // ---- Phase C: wq_eff = per-head q @ w_k^T (block-diag GEMM) -> WQE ------
  for (int nt = wave; nt < 80; nt += 8) {
    int h = nt / 20;
    int ntl = nt % 20;
    const __bf16* wk_h = wkT_f + (size_t)h * 102400;
    __builtin_prefetch(wk_h + (size_t)ntl * 512, 0, 1);
    v8f acc = {};
#pragma unroll 2
    for (int kt = 0; kt < 10; ++kt) {
      v16bf a = load_A(B1, HK_, h * KD_ + kt * 32, lane);
      v16bf b = load_Bfrag(wk_h, 20, kt, ntl, lane);
      acc = wmma_bf16(a, b, acc);
    }
    store_D_bf16(WQE, HK_, nt * 16, acc, lane);
  }
  __syncthreads();

  // ---- Phase D: logits S[r,h] = scale * C[r,:] . wq_eff[b,h,:] ------------
  const float scale = 0.05590169943749474f;  // 1/sqrt(320)
  for (int i = tid; i < ROWS_ * H_; i += THREADS_) {
    int r = i >> 2, h = i & 3;
    int lb = r / N_;
    const __bf16* crow = Cl + r * KD_;
    const __bf16* wrow = WQE + lb * HK_ + h * KD_;
    float s = 0.f;
#pragma unroll 8
    for (int c = 0; c < KD_; ++c) s += (float)crow[c] * (float)wrow[c];
    Sl[i] = s * scale;
  }
  __syncthreads();

  // ---- Phase E: softmax over N per (lb,h) ---------------------------------
  if (tid < TB_ * H_) {
    int lb = tid >> 2, h = tid & 3;
    float mx = -3.4e38f;
#pragma unroll
    for (int n = 0; n < N_; ++n) mx = fmaxf(mx, Sl[(lb * N_ + n) * H_ + h]);
    float e[N_], sum = 0.f;
#pragma unroll
    for (int n = 0; n < N_; ++n) {
      e[n] = __expf(Sl[(lb * N_ + n) * H_ + h] - mx);
      sum += e[n];
    }
    float inv = 1.f / sum;
#pragma unroll
    for (int n = 0; n < N_; ++n) Sl[(lb * N_ + n) * H_ + h] = e[n] * inv;
  }
  __syncthreads();

  // ---- Phase F: ctx[lb, h*320+c] = sum_n attn * C  -> B1 ------------------
  for (int i = tid; i < TB_ * HK_; i += THREADS_) {
    int lb = i / HK_, hc = i % HK_;
    int h = hc / KD_, c = hc % KD_;
    const int rb = lb * N_;
    float acc = 0.f;
#pragma unroll
    for (int n = 0; n < N_; ++n)
      acc += Sl[(rb + n) * H_ + h] * (float)Cl[(rb + n) * KD_ + c];
    B1[i] = (__bf16)acc;
  }
  __syncthreads();

  // ---- Phase G: x = ctx @ Wvf + fc_b + query (residual) -> XB f32 ---------
  for (int nt = wave; nt < 12; nt += 8) {
    v8f acc = {};
    for (int kt = 0; kt < 40; ++kt) {
      if ((kt & 7) == 0 && kt + 8 < 40)
        __builtin_prefetch(wvf_f + ((size_t)((kt + 8) * 12 + nt)) * 512, 0, 1);
      v16bf a = load_A(B1, HK_, kt * 32, lane);
      v16bf b = load_Bfrag(wvf_f, 12, kt, nt, lane);
      acc = wmma_bf16(a, b, acc);
    }
    int col = nt * 16 + (lane & 15);
    int mb = (lane >> 4) * 8;
    float fcb = fc_b[col];
#pragma unroll
    for (int v = 0; v < 8; ++v) {
      int m = mb + v;
      float qv = (col < E_) ? HD[m * E_ + col] : cosf(time_b[col - E_]);
      XB[m * QD_ + col] = acc[v] + fcb + qv;
    }
  }
  __syncthreads();

  // ---- Phase H: LayerNorm, build merge input HB = cat(h_dst, LN(x)) -------
  if (tid < TB_) {
    float mu = 0.f;
    for (int j = 0; j < QD_; ++j) mu += XB[tid * QD_ + j];
    mu *= (1.f / QD_);
    float var = 0.f;
    for (int j = 0; j < QD_; ++j) {
      float d = XB[tid * QD_ + j] - mu;
      var += d * d;
    }
    var *= (1.f / QD_);
    MU[tid] = mu;
    RS[tid] = rsqrtf(var + LN_EPS_);
  }
  __syncthreads();
  for (int i = tid; i < TB_ * KD_; i += THREADS_) {
    int m = i / KD_, j = i % KD_;
    float vsl;
    if (j < E_) {
      vsl = HD[m * E_ + j];
    } else {
      int jj = j - E_;
      vsl = ln_g[jj] * (XB[m * QD_ + jj] - MU[m]) * RS[m] + ln_b[jj];
    }
    HB[i] = (__bf16)vsl;
  }
  __syncthreads();

  // ---- Phase I: relu(HB @ mg1 + b1) -> Rl bf16   [16x320]@[320x128] -------
  {
    int nt = wave;  // 8 waves x 1 tile == 128 cols
    __builtin_prefetch(mg1_f + (size_t)nt * 512, 0, 1);
    v8f acc = {};
#pragma unroll 2
    for (int kt = 0; kt < 10; ++kt) {
      v16bf a = load_A(HB, KD_, kt * 32, lane);
      v16bf b = load_Bfrag(mg1_f, 8, kt, nt, lane);
      acc = wmma_bf16(a, b, acc);
    }
    int col = nt * 16 + (lane & 15);
    int mb = (lane >> 4) * 8;
    float bb = mg1_b[col];
#pragma unroll
    for (int v = 0; v < 8; ++v)
      Rl[(mb + v) * E_ + col] = (__bf16)fmaxf(acc[v] + bb, 0.f);
  }
  __syncthreads();

  // ---- Phase J: out = Rl @ mg2 + b2 -> global f32  [16x128]@[128x128] -----
  {
    int nt = wave;
    v8f acc = {};
#pragma unroll
    for (int kt = 0; kt < 4; ++kt) {
      v16bf a = load_A(Rl, E_, kt * 32, lane);
      v16bf b = load_Bfrag(mg2_f, 8, kt, nt, lane);
      acc = wmma_bf16(a, b, acc);
    }
    int col = nt * 16 + (lane & 15);
    int mb = (lane >> 4) * 8;
    float bb = mg2_b[col];
#pragma unroll
    for (int v = 0; v < 8; ++v)
      out[((size_t)(b0 + mb + v)) * E_ + col] = acc[v] + bb;
  }
}

// ---------------------------------------------------------------------------
extern "C" void kernel_launch(void* const* d_in, const int* in_sizes, int n_in,
                              void* d_out, int out_size, void* d_ws,
                              size_t ws_size, hipStream_t stream) {
  const float* h_dst     = (const float*)d_in[0];
  const float* src_h     = (const float*)d_in[1];
  const float* edge_feat = (const float*)d_in[2];
  const float* delta_t   = (const float*)d_in[3];
  const float* time_w    = (const float*)d_in[4];
  const float* time_b    = (const float*)d_in[5];
  const float* w_q       = (const float*)d_in[6];
  const float* w_k       = (const float*)d_in[7];
  const float* w_v       = (const float*)d_in[8];
  const float* fc_w      = (const float*)d_in[9];
  const float* fc_b      = (const float*)d_in[10];
  const float* ln_g      = (const float*)d_in[11];
  const float* ln_b      = (const float*)d_in[12];
  const float* mg1_w     = (const float*)d_in[13];
  const float* mg1_b     = (const float*)d_in[14];
  const float* mg2_w     = (const float*)d_in[15];
  const float* mg2_b     = (const float*)d_in[16];
  const int*   nbr_idx   = (const int*)d_in[17];
  (void)in_sizes; (void)n_in; (void)out_size; (void)ws_size;

  // bf16 fragment-ready weight staging in workspace (~1.9 MB, L2-resident)
  char* ws = (char*)d_ws;
  __bf16* wq_f  = (__bf16*)(ws + 0);        // 491520 B
  __bf16* wkT_f = (__bf16*)(ws + 491520);   // 819200 B
  __bf16* wvf_f = (__bf16*)(ws + 1310720);  // 491520 B
  __bf16* mg1_f = (__bf16*)(ws + 1802240);  //  81920 B
  __bf16* mg2_f = (__bf16*)(ws + 1884160);  //  32768 B

  prep_weights<<<(PW_T3 + 255) / 256, 256, 0, stream>>>(
      w_q, w_k, mg1_w, mg2_w, wq_f, wkT_f, mg1_f, mg2_f);
  prep_wvf<<<(HK_ * QD_ + 255) / 256, 256, 0, stream>>>(w_v, fc_w, wvf_f);

  (void)hipFuncSetAttribute((const void*)attn_fused,
                            hipFuncAttributeMaxDynamicSharedMemorySize, SMEM_);
  attn_fused<<<B_ / TB_, THREADS_, SMEM_, stream>>>(
      h_dst, src_h, edge_feat, delta_t, time_w, time_b, fc_b, ln_g, ln_b,
      mg1_b, mg2_b, nbr_idx, wq_f, wkT_f, wvf_f, mg1_f, mg2_f,
      (float*)d_out);
}
